// NATTENCrossAttention_16157666968362
// MI455X (gfx1250) — matrix operand order repaired and checked by reading, not compiled, not run on hardware
//
#include <hip/hip_runtime.h>

// ---------- CDNA5 WMMA types ----------
typedef __attribute__((ext_vector_type(16))) __bf16 v16bf;
typedef __attribute__((ext_vector_type(8)))  float  v8f;
typedef __attribute__((ext_vector_type(4)))  __bf16 v4bf;

union FragU { struct { uint4 lo, hi; } u; v16bf v; };

// A-matrix 16x32 bf16 fragment from a 64-byte LDS span (one row of 32 bf16).
// Lane kh = lane>>4:  VGPR0-3 <- K[8kh..8kh+7]  (bytes kh*16 .. +16)
//                     VGPR4-7 <- K[16+8kh .. ]  (bytes 32+kh*16 .. +16)
__device__ __forceinline__ v16bf fragA64(const void* span, int kh) {
  const char* s = (const char*)span;
  FragU f;
  f.u.lo = *(const uint4*)(s + kh * 16);
  f.u.hi = *(const uint4*)(s + 32 + kh * 16);
  return f.v;
}
// B-matrix 32x16 bf16 fragment from a 64-byte LDS span (one column of 32 bf16).
// Lane kh: VGPR0-3 <- K[8kh..8kh+7] (bytes kh*16), VGPR4-7 <- K[16+8kh ..] (bytes 32+kh*16)
__device__ __forceinline__ v16bf fragB64(const void* span, int kh) {
  const char* s = (const char*)span;
  FragU f;
  f.u.lo = *(const uint4*)(s + kh * 16);
  f.u.hi = *(const uint4*)(s + 32 + kh * 16);
  return f.v;
}
__device__ __forceinline__ v8f wmma_bf16(v16bf a, v16bf b, v8f c) {
  return __builtin_amdgcn_wmma_f32_16x16x32_bf16(false, a, false, b, (short)0, c,
                                                 false, false);
}
__device__ __forceinline__ int iclamp(int x, int lo, int hi) {
  x = x < lo ? lo : x;
  return x > hi ? hi : x;
}
__device__ __forceinline__ v4bf pack4(float a, float b, float c, float d) {
  v4bf p;
  p[0] = (__bf16)a; p[1] = (__bf16)b; p[2] = (__bf16)c; p[3] = (__bf16)d;
  return p;
}

// ---------- Stage 1: Y(256 x Np) = W(256x256) * X(256 x Np) + bias ----------
// Block: 256 threads (8 waves), tile 32(M) x 256(N), K-steps of 32.
__global__ __launch_bounds__(256) void gemm_bias_kernel(
    const float* __restrict__ Wm, const float* __restrict__ X,
    const float* __restrict__ bias, float* __restrict__ Y, int Np) {
  __shared__ alignas(16) __bf16 ldsA[32 * 32];   // row-major, 64B per row
  __shared__ alignas(16) __bf16 ldsB[256 * 32];  // col-major, 64B per column

  const int t = threadIdx.x;
  const int lane = t & 31, wave = t >> 5;
  const int m0 = blockIdx.x * 32;
  const int n0 = blockIdx.y * 256;
  const int mi = wave & 1;          // which 16-row half of the 32-row tile
  const int ntb = (wave >> 1) * 4;  // base of this wave's 4 N-tiles
  const int kh = lane >> 4;

  v8f acc[4];
#pragma unroll
  for (int q = 0; q < 4; ++q) acc[q] = (v8f){};

  for (int k0 = 0; k0 < 256; k0 += 32) {
    // Prefetch next K-step tiles (L2-resident problem; gfx1250 global_prefetch)
    if (k0 + 32 < 256) {
      __builtin_prefetch(&X[(k0 + 32) * Np + n0 + t], 0, 0);
      __builtin_prefetch(&Wm[(m0 + (t >> 3)) * 256 + (k0 + 32) + (t & 7) * 4], 0, 0);
    }
    // A tile 32x32: 4 floats per thread, coalesced float4 loads
    {
      int row = t >> 3;
      int col = (t & 7) * 4;
      const float4 wv = *(const float4*)(&Wm[(m0 + row) * 256 + k0 + col]);
      *(v4bf*)(&ldsA[row * 32 + col]) = pack4(wv.x, wv.y, wv.z, wv.w);
    }
    // B tile 32 k x 256 n, column-major in LDS; pack 4 K-rows per ds_store_b64
    const float* xcol = &X[k0 * Np + n0 + t];
#pragma unroll
    for (int r = 0; r < 32; r += 4) {
      float a0 = xcol[(r + 0) * Np];
      float a1 = xcol[(r + 1) * Np];
      float a2 = xcol[(r + 2) * Np];
      float a3 = xcol[(r + 3) * Np];
      *(v4bf*)(&ldsB[t * 32 + r]) = pack4(a0, a1, a2, a3);
    }
    __syncthreads();

    v16bf afrag = fragA64(&ldsA[(mi * 16 + (lane & 15)) * 32], kh);
#pragma unroll
    for (int q = 0; q < 4; ++q) {
      int cloc = (ntb + q) * 16 + (lane & 15);
      v16bf bfrag = fragB64(&ldsB[cloc * 32], kh);
      acc[q] = wmma_bf16(afrag, bfrag, acc[q]);
    }
    __syncthreads();
  }
  // D layout: VGPR r -> row (r + 8*kh), col (lane&15)
#pragma unroll
  for (int q = 0; q < 4; ++q) {
    int ncol = n0 + (ntb + q) * 16 + (lane & 15);
#pragma unroll
    for (int r = 0; r < 8; ++r) {
      int m = m0 + mi * 16 + r + 8 * kh;
      Y[m * Np + ncol] = acc[q][r] + bias[m];
    }
  }
}

// ---------- Stage 2: NATTEN attention ----------
// One block (128 thr / 4 waves) per 4x4 output tile. Dilation-2 + exact-2x
// upsample collapse: output pixel x reads input rows clamp(x/2-3,0,41)+j on
// the 48-grid, so the tile's neighbor superset is an 8x8 patch (64 pixels).
__global__ __launch_bounds__(128) void natten_attn_kernel(
    const float* __restrict__ qp, const float* __restrict__ kp,
    const float* __restrict__ v, float* __restrict__ out) {
  __shared__ alignas(16) __bf16 ldsQ[8 * 16 * 32];  // [head][px][ch]   8 KB
  __shared__ alignas(16) __bf16 ldsK[8 * 64 * 32];  // [head][nbr][ch] 32 KB
  __shared__ alignas(16) __bf16 ldsV[128 * 64];     // [cv][nbr]       16 KB
  __shared__ alignas(16) float  ldsL[8 * 16 * 64];  // logits/probs    32 KB
  __shared__ alignas(16) __bf16 ldsP[16 * 64];      // head-mean probs  2 KB

  const int t = threadIdx.x;
  const int lane = t & 31, wave = t >> 5;
  const int kh = lane >> 4;
  const int x0 = blockIdx.x * 4, y0 = blockIdx.y * 4;
  const int R0 = iclamp(x0 / 2 - 3, 0, 41);
  const int C0 = iclamp(y0 / 2 - 3, 0, 41);
  const float scale = 0.17677669529663687f;  // 1/sqrt(32)

  // --- stage Q (scaled): 4 channels per ds_store_b64 ---
  for (int e = t; e < 1024; e += 128) {
    int c4 = (e & 7) * 4, px = (e >> 3) & 15, h = e >> 7;
    int p = (x0 + (px >> 2)) * 96 + (y0 + (px & 3));
    const float* src = &qp[(h * 32 + c4) * 9216 + p];
    *(v4bf*)(&ldsQ[(h * 16 + px) * 32 + c4]) =
        pack4(src[0] * scale, src[9216] * scale, src[2 * 9216] * scale,
              src[3 * 9216] * scale);
  }
  // --- stage K: 4 channels per ds_store_b64 ---
  for (int e = t; e < 4096; e += 128) {
    int nbr = e & 63, c4 = (e >> 6) * 4;  // c4 in 0..252
    int pix = (R0 + (nbr >> 3)) * 48 + (C0 + (nbr & 7));
    const float* src = &kp[c4 * 2304 + pix];
    int h = c4 >> 5, ch = c4 & 31;
    *(v4bf*)(&ldsK[(h * 64 + nbr) * 32 + ch]) =
        pack4(src[0], src[2304], src[2 * 2304], src[3 * 2304]);
  }
  // --- stage V: 4 consecutive neighbors (same 8-wide row) per ds_store_b64 ---
  for (int e = t; e < 2048; e += 128) {
    int n4 = (e & 15) * 4;  // 0..60, stays within one row of 8
    int cv = e >> 4;        // 0..127
    int pix = (R0 + (n4 >> 3)) * 48 + (C0 + (n4 & 7));
    const float* src = &v[cv * 2304 + pix];
    *(v4bf*)(&ldsV[cv * 64 + n4]) = pack4(src[0], src[1], src[2], src[3]);
  }
  __syncthreads();

  // --- logits = q . k over head dim (K=32): 2 heads per wave, 4 n-tiles ---
#pragma unroll
  for (int hh = 0; hh < 2; ++hh) {
    int h = wave * 2 + hh;
    v16bf afrag = fragA64(&ldsQ[(h * 16 + (lane & 15)) * 32], kh);
#pragma unroll
    for (int nt = 0; nt < 4; ++nt) {
      v16bf bfrag = fragB64(&ldsK[(h * 64 + nt * 16 + (lane & 15)) * 32], kh);
      v8f accl = (v8f){};
      accl = wmma_bf16(afrag, bfrag, accl);
      int col = nt * 16 + (lane & 15);
      int u = col >> 3, vv = col & 7;
#pragma unroll
      for (int r = 0; r < 8; ++r) {
        int px = r + 8 * kh;
        int ox = iclamp((x0 + (px >> 2)) / 2 - 3, 0, 41) - R0;
        int oy = iclamp((y0 + (px & 3)) / 2 - 3, 0, 41) - C0;
        bool valid = (u >= ox) && (u <= ox + 6) && (vv >= oy) && (vv <= oy + 6);
        ldsL[(h * 16 + px) * 64 + col] = valid ? accl[r] : -1e30f;
      }
    }
  }
  __syncthreads();

  // --- softmax: one thread per (head, pixel) row of 64 ---
  {
    int h = t >> 4, px = t & 15;
    float* row = &ldsL[(h * 16 + px) * 64];
    float mx = -1e30f;
    for (int i = 0; i < 64; ++i) mx = fmaxf(mx, row[i]);
    float sum = 0.f;
    for (int i = 0; i < 64; ++i) {
      float e = __expf(row[i] - mx);
      row[i] = e;
      sum += e;
    }
    float inv = 1.f / sum;
    for (int i = 0; i < 64; ++i) row[i] *= inv;
  }
  __syncthreads();

  // --- mean over heads -> bf16 P (16 x 64) ---
  for (int e = t; e < 1024; e += 128) {
    int px = e >> 6, col = e & 63;
    float s = 0.f;
#pragma unroll
    for (int h = 0; h < 8; ++h) s += ldsL[(h * 16 + px) * 64 + col];
    ldsP[px * 64 + col] = (__bf16)(s * 0.125f);
  }
  __syncthreads();

  // --- out(16 px x 128 cv) = P(16x64) @ V(64x128): 2 cv-tiles per wave ---
  // Stage D into LDS (reuse ldsL) so the global store is pixel-coalesced.
  float* ldsO = ldsL;  // [px][cv] = 16 x 128 floats (8 KB of the 32 KB buffer)
#pragma unroll
  for (int q = 0; q < 2; ++q) {
    int ncv = (wave * 2 + q) * 16;
    v8f acco = (v8f){};
#pragma unroll
    for (int kf = 0; kf < 2; ++kf) {
      v16bf afrag = fragA64(&ldsP[(lane & 15) * 64 + kf * 32], kh);
      v16bf bfrag = fragB64(&ldsV[(ncv + (lane & 15)) * 64 + kf * 32], kh);
      acco = wmma_bf16(afrag, bfrag, acco);
    }
    int cv = ncv + (lane & 15);
#pragma unroll
    for (int r = 0; r < 8; ++r) {
      int px = r + 8 * kh;
      ldsO[px * 128 + cv] = acco[r];
    }
  }
  __syncthreads();

  // coalesced store: consecutive threads -> consecutive pixels for fixed cv
  for (int e = t; e < 2048; e += 128) {
    int cv = e >> 4, px = e & 15;
    int p = (x0 + (px >> 2)) * 96 + (y0 + (px & 3));
    out[cv * 9216 + p] = ldsO[px * 128 + cv];
  }
}

// ---------- host launcher ----------
extern "C" void kernel_launch(void* const* d_in, const int* in_sizes, int n_in,
                              void* d_out, int out_size, void* d_ws, size_t ws_size,
                              hipStream_t stream) {
  const float* q   = (const float*)d_in[0];  // (1,256,96,96)
  const float* k   = (const float*)d_in[1];  // (1,256,48,48)
  const float* v   = (const float*)d_in[2];  // (1,128,48,48)
  const float* w_q = (const float*)d_in[3];  // (256,256)
  const float* b_q = (const float*)d_in[4];  // (256,)
  const float* w_k = (const float*)d_in[5];  // (256,256)
  const float* b_k = (const float*)d_in[6];  // (256,)
  // d_in[7]=kernel_size(7), d_in[8]=dilation(2): specialized at compile time.

  float* qp = (float*)d_ws;            // 256*9216 f32
  float* kp = qp + 256 * 9216;         // 256*2304 f32

  gemm_bias_kernel<<<dim3(8, 36), 256, 0, stream>>>(w_q, q, b_q, qp, 9216);
  gemm_bias_kernel<<<dim3(8, 9),  256, 0, stream>>>(w_k, k, b_k, kp, 2304);
  natten_attn_kernel<<<dim3(24, 24), 128, 0, stream>>>(qp, kp, v, (float*)d_out);
}
